// WindowAttention_82858509074994
// MI455X (gfx1250) — matrix operand order repaired and checked by reading, not compiled, not run on hardware
//
#include <hip/hip_runtime.h>

// Windowed 3D attention (Swin-style), fully fused per window.
// DIM=96, HEADS=3, hd=32, N=343 (pad to 352 = 22*16), B=1024 windows.
// Memory-bound (~270MB HBM traffic, ~12us floor @ 23.3TB/s), so: one block per
// window, everything resident in LDS, f16 WMMA (f32 accum) for all GEMMs.

#define DIMC 96
#define NH   3
#define NTOK 343
#define NT   352      // 22 tiles of 16
#define NKT  22
#define QKV3 288

typedef __attribute__((ext_vector_type(16))) _Float16 v16h;
typedef __attribute__((ext_vector_type(8)))  _Float16 h8;
typedef __attribute__((ext_vector_type(4)))  _Float16 h4;
typedef __attribute__((ext_vector_type(8)))  float    v8f;

__device__ __forceinline__ v16h cat16(h8 lo, h8 hi) {
  v16h r;
#pragma unroll
  for (int i = 0; i < 8; ++i) { r[i] = lo[i]; r[i + 8] = hi[i]; }
  return r;
}

__device__ __forceinline__ v8f wmma_f16(v16h a, v16h b, v8f c) {
  return __builtin_amdgcn_wmma_f32_16x16x32_f16(false, a, false, b, (short)0, c,
                                                false, false);
}

// ---------------- prep: transpose weights to f16 (B-operand friendly) -------
__global__ void prep_weights_kernel(const float* __restrict__ qkv_w,
                                    const float* __restrict__ proj_w,
                                    _Float16* __restrict__ qkvT,
                                    _Float16* __restrict__ projT) {
  int idx = blockIdx.x * 256 + threadIdx.x;
  if (idx < QKV3 * DIMC) {
    int c = idx / DIMC, k = idx % DIMC;
    qkvT[idx] = (_Float16)qkv_w[k * QKV3 + c];          // qkvT[c][k]
  } else if (idx < QKV3 * DIMC + DIMC * DIMC) {
    int j = idx - QKV3 * DIMC;
    int c = j / DIMC, k = j % DIMC;
    projT[j] = (_Float16)proj_w[k * DIMC + c];          // projT[c][k]
  }
}

// prep: expand rel-pos bias (+mask) into transposed padded f16: biasT[h][m][n]
__global__ void prep_bias_kernel(const float* __restrict__ bias_table,
                                 const float* __restrict__ mask,
                                 const int* __restrict__ rel_index,
                                 _Float16* __restrict__ biasT) {
  int idx = blockIdx.x * 256 + threadIdx.x;
  if (idx >= NH * NT * NT) return;
  int h = idx / (NT * NT);
  int rem = idx % (NT * NT);
  int m = rem / NT;   // key token
  int n = rem % NT;   // query token
  float v = 0.f;
  if (m < NTOK && n < NTOK)
    v = bias_table[rel_index[n * NTOK + m] * NH + h] + mask[n * NTOK + m];
  biasT[idx] = (_Float16)v;
}

// ---------------- main fused kernel: one block per window -------------------
__global__ __launch_bounds__(256, 1)
void win_attn_kernel(const float* __restrict__ x,
                     const float* __restrict__ qkv_b,
                     const float* __restrict__ proj_b,
                     const _Float16* __restrict__ qkvT,
                     const _Float16* __restrict__ projT,
                     const _Float16* __restrict__ biasT,
                     float* __restrict__ out) {
  __shared__ __attribute__((aligned(64))) _Float16 Xs[NT * DIMC];      // 66KB, reused as O
  __shared__ __attribute__((aligned(64))) _Float16 Qs[NH * NT * 32];   // 66KB
  __shared__ __attribute__((aligned(64))) _Float16 Ks[NH * NT * 32];   // 66KB
  __shared__ __attribute__((aligned(64))) _Float16 Vt[NH * 32 * NT];   // 66KB (transposed)
  __shared__ __attribute__((aligned(64))) _Float16 Pst[8 * 16 * 32];   // 8KB (per-wave P stage)

  const int tid  = threadIdx.x;
  const int wave = tid >> 5;
  const int lane = tid & 31;
  const int l16  = lane & 15;
  const int hl   = lane >> 4;
  const int blk  = blockIdx.x;

  // ---- Phase 0: stage x -> LDS f16, zero-pad rows 343..351 ----
  const float* xg = x + (size_t)blk * NTOK * DIMC;
  for (int idx = tid; idx < NTOK * (DIMC / 4); idx += 256) {
    float4 v = ((const float4*)xg)[idx];
    h4 hv; hv[0] = (_Float16)v.x; hv[1] = (_Float16)v.y;
    hv[2] = (_Float16)v.z; hv[3] = (_Float16)v.w;
    *(h4*)&Xs[idx * 4] = hv;
  }
  for (int idx = tid; idx < (NT - NTOK) * (DIMC / 4); idx += 256) {
    h4 z = {};
    *(h4*)&Xs[NTOK * DIMC + idx * 4] = z;
  }
  __syncthreads();

  // ---- Phase 1: QKV = Xs @ qkv_w + b ; scatter into Q / K / V^T layouts ----
  for (int u = wave; u < NKT * 18; u += 8) {
    int ti = u % NKT, tc = u / NKT;
    int gcol = tc * 16 + l16;
    v8f acc = {};
#pragma unroll
    for (int kk = 0; kk < 3; ++kk) {
      int arow = ti * 16 + l16;
      int ak = kk * 32 + hl * 16;
      v16h A = cat16(*(const h8*)&Xs[arow * DIMC + ak],
                     *(const h8*)&Xs[arow * DIMC + ak + 8]);
      v16h Bv = *(const v16h*)&qkvT[gcol * DIMC + kk * 32 + hl * 16];
      acc = wmma_f16(A, Bv, acc);
    }
    float bc = qkv_b[gcol];
    int s = gcol / DIMC; int w = gcol % DIMC; int h = w >> 5; int ch = w & 31;
    int rbase = ti * 16 + hl * 8;
    if (s == 0) {
      const float sc = 0.17677669529663687f;   // hd^-0.5
#pragma unroll
      for (int r = 0; r < 8; ++r)
        Qs[(h * NT + rbase + r) * 32 + ch] = (_Float16)((acc[r] + bc) * sc);
    } else if (s == 1) {
#pragma unroll
      for (int r = 0; r < 8; ++r)
        Ks[(h * NT + rbase + r) * 32 + ch] = (_Float16)(acc[r] + bc);
    } else {
      h8 vv;
#pragma unroll
      for (int r = 0; r < 8; ++r) vv[r] = (_Float16)(acc[r] + bc);
      *(h8*)&Vt[(h * 32 + ch) * NT + rbase] = vv;   // V transposed: [h][ch][tok]
    }
  }
  __syncthreads();

  // ---- Phase 2: flash attention, one (query-tile, head) unit per wave ----
  _Float16* pst = &Pst[wave * 512];
  for (int u = wave; u < NH * NKT; u += 8) {
    int i = u % NKT; int h = u / NKT;
    int qrow = i * 16 + l16;
    int k0 = hl * 16;
    v16h aQ = cat16(*(const h8*)&Qs[(h * NT + qrow) * 32 + k0],
                    *(const h8*)&Qs[(h * NT + qrow) * 32 + k0 + 8]);
    v8f o0 = {}, o1 = {};
    float mrow[8], lrow[8];
#pragma unroll
    for (int r = 0; r < 8; ++r) { mrow[r] = -3.0e38f; lrow[r] = 0.f; }

    for (int t = 0; t < 11; ++t) {                 // 11 pairs of 16-key tiles
      int tok0 = t * 32 + l16;
      int tok1 = tok0 + 16;
      v16h bk0 = *(const v16h*)&Ks[(h * NT + tok0) * 32 + hl * 16];
      v16h bk1 = *(const v16h*)&Ks[(h * NT + tok1) * 32 + hl * 16];
      v8f z = {};
      v8f s0 = wmma_f16(aQ, bk0, z);
      v8f s1 = wmma_f16(aQ, bk1, z);
      h8 b0 = *(const h8*)&biasT[((size_t)(h * NT + tok0)) * NT + i * 16 + hl * 8];
      h8 b1 = *(const h8*)&biasT[((size_t)(h * NT + tok1)) * NT + i * 16 + hl * 8];
      bool va = tok0 < NTOK, vb = tok1 < NTOK;
      float p0[8], p1[8], cm[8];
#pragma unroll
      for (int r = 0; r < 8; ++r) {
        float x0 = va ? s0[r] + (float)b0[r] : -1.0e30f;
        float x1 = vb ? s1[r] + (float)b1[r] : -1.0e30f;
        p0[r] = x0; p1[r] = x1; cm[r] = fmaxf(x0, x1);
      }
#pragma unroll
      for (int r = 0; r < 8; ++r) {
        cm[r] = fmaxf(cm[r], __shfl_xor(cm[r], 1, 16));
        cm[r] = fmaxf(cm[r], __shfl_xor(cm[r], 2, 16));
        cm[r] = fmaxf(cm[r], __shfl_xor(cm[r], 4, 16));
        cm[r] = fmaxf(cm[r], __shfl_xor(cm[r], 8, 16));
        float mn = fmaxf(mrow[r], cm[r]);
        float f  = __expf(mrow[r] - mn);
        mrow[r]  = mn;
        p0[r] = __expf(p0[r] - mn);
        p1[r] = __expf(p1[r] - mn);
        float rs = p0[r] + p1[r];
        rs += __shfl_xor(rs, 1, 16);
        rs += __shfl_xor(rs, 2, 16);
        rs += __shfl_xor(rs, 4, 16);
        rs += __shfl_xor(rs, 8, 16);
        lrow[r] = lrow[r] * f + rs;
        o0[r] *= f; o1[r] *= f;
      }
      // stage P (D-layout -> A-layout) through per-wave LDS scratch
#pragma unroll
      for (int r = 0; r < 8; ++r) {
        pst[(hl * 8 + r) * 32 + l16]      = (_Float16)p0[r];
        pst[(hl * 8 + r) * 32 + 16 + l16] = (_Float16)p1[r];
      }
      asm volatile("s_wait_dscnt 0" ::: "memory");   // wave-local LDS RAW fence
      v16h aP = cat16(*(const h8*)&pst[l16 * 32 + hl * 16],
                      *(const h8*)&pst[l16 * 32 + hl * 16 + 8]);
      v16h bv0 = *(const v16h*)&Vt[(h * 32 + l16) * NT + t * 32 + hl * 16];
      v16h bv1 = *(const v16h*)&Vt[(h * 32 + 16 + l16) * NT + t * 32 + hl * 16];
      o0 = wmma_f16(aP, bv0, o0);
      o1 = wmma_f16(aP, bv1, o1);
    }
    // normalize and store O into Xs (reused as O buffer)
#pragma unroll
    for (int r = 0; r < 8; ++r) {
      float inv = 1.0f / lrow[r];
      int gr = i * 16 + hl * 8 + r;
      Xs[gr * DIMC + h * 32 + l16]      = (_Float16)(o0[r] * inv);
      Xs[gr * DIMC + h * 32 + 16 + l16] = (_Float16)(o1[r] * inv);
    }
  }
  __syncthreads();

  // ---- Phase 3: out = O @ proj_w + proj_b ----
  float* og = out + (size_t)blk * NTOK * DIMC;
  for (int u = wave; u < NKT * 6; u += 8) {
    int ti = u % NKT, tc = u / NKT;
    int gc = tc * 16 + l16;
    v8f acc = {};
#pragma unroll
    for (int kk = 0; kk < 3; ++kk) {
      int arow = ti * 16 + l16;
      int ak = kk * 32 + hl * 16;
      v16h A = cat16(*(const h8*)&Xs[arow * DIMC + ak],
                     *(const h8*)&Xs[arow * DIMC + ak + 8]);
      v16h Bv = *(const v16h*)&projT[gc * DIMC + kk * 32 + hl * 16];
      acc = wmma_f16(A, Bv, acc);
    }
    float bb = proj_b[gc];
#pragma unroll
    for (int r = 0; r < 8; ++r) {
      int gr = ti * 16 + hl * 8 + r;
      if (gr < NTOK) og[gr * DIMC + gc] = acc[r] + bb;
    }
  }
}

extern "C" void kernel_launch(void* const* d_in, const int* in_sizes, int n_in,
                              void* d_out, int out_size, void* d_ws, size_t ws_size,
                              hipStream_t stream) {
  const float* x        = (const float*)d_in[0];
  const float* qkv_w    = (const float*)d_in[1];
  const float* qkv_b    = (const float*)d_in[2];
  const float* proj_w   = (const float*)d_in[3];
  const float* proj_b   = (const float*)d_in[4];
  const float* bias_tab = (const float*)d_in[5];
  const float* mask     = (const float*)d_in[6];
  const int*   rel_idx  = (const int*)d_in[7];
  float* out = (float*)d_out;

  // workspace layout (f16): qkvT[288*96] | projT[96*96] | biasT[3*352*352]
  _Float16* qkvT  = (_Float16*)d_ws;
  _Float16* projT = qkvT + QKV3 * DIMC;
  _Float16* biasT = projT + DIMC * DIMC;

  int nb = in_sizes[0] / (NTOK * DIMC);   // number of windows (1024)

  prep_weights_kernel<<<(QKV3 * DIMC + DIMC * DIMC + 255) / 256, 256, 0, stream>>>(
      qkv_w, proj_w, qkvT, projT);
  prep_bias_kernel<<<(NH * NT * NT + 255) / 256, 256, 0, stream>>>(
      bias_tab, mask, rel_idx, biasT);
  win_attn_kernel<<<nb, 256, 0, stream>>>(x, qkv_b, proj_b, qkvT, projT, biasT, out);
}
